// SGCRN_32581621907810
// MI455X (gfx1250) — compile-verified, matrix operand order checked by reading, not run on hardware
//
#include <hip/hip_runtime.h>
#include <hip/hip_bf16.h>

// ---------------------------------------------------------------------------
// SGCRN on MI455X (gfx1250): bf16 WMMA for both heavy GEMMs, f32 accumulate.
//  * gates are independent of the recurrence -> message-passing GEMM batched
//    over all 12 timesteps in ONE launch per layer; cheap sequential cell scan
//  * dynamic layers 1 and 2 are identical -> computed once, written twice
//  * GEMM inner loop: branch-free (clamped B loads), 2-stage software pipeline,
//    loads written directly off kernel args so they lower to global_load_b128
//    (not flat_*, which would also tick DScnt).
// ---------------------------------------------------------------------------

typedef __bf16 bf16;
typedef __attribute__((ext_vector_type(16))) __bf16 v16bf;
typedef __attribute__((ext_vector_type(8)))  __bf16 v8bf;
typedef __attribute__((ext_vector_type(8)))  float  v8f;

#define NN   2048      // nodes
#define BB   16        // batch
#define LL   12        // timesteps
#define MM   112       // model size
#define CC   (BB*MM)   // 1792 columns per timestep of the message-passing GEMM
#define CCL  (LL*CC)   // 21504 columns batched over all timesteps
#define KPAD 128       // M padded to a multiple of 32 for the gates GEMM
#define G4M  448       // 4*M gate width

__device__ inline v16bf cat16(v8bf lo, v8bf hi) {
  return __builtin_shufflevector(lo, hi, 0,1,2,3,4,5,6,7,8,9,10,11,12,13,14,15);
}
__device__ inline v16bf ldfrag(const bf16* __restrict__ p) {
  return cat16(*(const v8bf*)p, *(const v8bf*)(p + 16));
}
// gfx1250 global prefetch: no counter traffic, stays on the global path.
__device__ inline void gprefetch(const bf16* p) {
  asm volatile("global_prefetch_b8 %0, off scope:SCOPE_SE" :: "v"(p));
}

// ---------------------------------------------------------------------------
// bf16 GEMM, C = A(row-major, [Mrows x K]) * B^T(row-major, [Ncols x K]).
// Block tile 128x128, 8 waves (2x4), wave tile 64x32 -> 8 v8f accumulators.
// B loads are clamped to the last valid column (branch-free inner loop);
// out-of-range results are discarded by the epilogue guard.
// MODE 0: write f32 C (row-major, ldc)                         -> gates
// MODE 1: batched xa epilogue, col c = t*1792 + b*112 + m, write bf16 into
//         Cx[((t*BB + b)*NN + row)*KPAD + m]  (A-operand of the gates GEMM)
// ---------------------------------------------------------------------------
template<int MODE>
__global__ __launch_bounds__(256)
void gemm_wmma(const bf16* __restrict__ A, const bf16* __restrict__ BT,
               float* __restrict__ Cf, bf16* __restrict__ Cx,
               int Ncols, int K, int lda, int ldb, int ldc)
{
  const int lane = threadIdx.x & 31;
  const int w    = threadIdx.x >> 5;
  const int wrow = w & 1, wcol = w >> 1;
  const int row0 = blockIdx.x * 128 + wrow * 64;
  const int col0 = blockIdx.y * 128 + wcol * 32;
  const int hsel = (lane < 16) ? 0 : 8;   // ISA 16-bit operand K-half select
  const int l15  = lane & 15;

  v8f acc[4][2];
#pragma unroll
  for (int i = 0; i < 4; ++i)
#pragma unroll
    for (int j = 0; j < 2; ++j)
#pragma unroll
      for (int e = 0; e < 8; ++e) acc[i][j][e] = 0.f;

  // scalar element offsets; all loads stay rooted at the kernel arguments
  size_t aoff[4];
#pragma unroll
  for (int rt = 0; rt < 4; ++rt)
    aoff[rt] = (size_t)(row0 + rt * 16 + l15) * lda + hsel;
  size_t boff[2];
#pragma unroll
  for (int ct = 0; ct < 2; ++ct) {
    int col = col0 + ct * 16 + l15;
    col = (col < Ncols) ? col : (Ncols - 1);   // clamp: OOB cols discarded later
    boff[ct] = (size_t)col * ldb + hsel;
  }

  auto ldA = [&](int kk, v16bf (&af)[4]) {
#pragma unroll
    for (int rt = 0; rt < 4; ++rt) af[rt] = ldfrag(A + aoff[rt] + kk);
  };
  auto ldB = [&](int kk, v16bf (&bv)[2]) {
#pragma unroll
    for (int ct = 0; ct < 2; ++ct) bv[ct] = ldfrag(BT + boff[ct] + kk);
  };
  auto wmma8 = [&](const v16bf (&af)[4], const v16bf (&bv)[2]) {
#pragma unroll
    for (int rt = 0; rt < 4; ++rt)
#pragma unroll
      for (int ct = 0; ct < 2; ++ct)
        acc[rt][ct] = __builtin_amdgcn_wmma_f32_16x16x32_bf16(
            false, af[rt], false, bv[ct], (short)0, acc[rt][ct], false, false);
  };

  v16bf a0[4], b0[2], a1[4], b1[2];
  ldA(0, a0); ldB(0, b0);                       // chunk 0 in flight

  int kk = 0;
  for (; kk + 64 < K; kk += 64) {
    gprefetch(A + aoff[0] + kk + 512);          // ~1KB ahead on each stream
    gprefetch(BT + boff[0] + kk + 512);
    ldA(kk + 32, a1); ldB(kk + 32, b1);         // next chunk in flight
    wmma8(a0, b0);                              // compute current chunk
    ldA(kk + 64, a0); ldB(kk + 64, b0);         // chunk after next in flight
    wmma8(a1, b1);
  }
  ldA(K - 32, a1); ldB(K - 32, b1);             // last chunk
  wmma8(a0, b0);
  wmma8(a1, b1);

  // Epilogue. C/D layout: VGPR v holds row (v) for lanes 0-15, (v+8) for 16-31;
  // column = lane&15.
#pragma unroll
  for (int rt = 0; rt < 4; ++rt) {
    const int rbase = row0 + rt * 16 + ((lane >= 16) ? 8 : 0);
#pragma unroll
    for (int ct = 0; ct < 2; ++ct) {
      const int c = col0 + ct * 16 + l15;
      if (MODE == 0) {
        if (c < Ncols) {
#pragma unroll
          for (int v = 0; v < 8; ++v)
            Cf[(size_t)(rbase + v) * ldc + c] = acc[rt][ct][v];
        }
      } else {
        const int t  = c / CC;          // timestep
        const int r2 = c - t * CC;
        const int bq = r2 / MM;         // batch
        const int mq = r2 - bq * MM;    // feature
        bf16* dst = Cx + (((size_t)t * BB + bq) * NN + rbase) * KPAD + mq;
#pragma unroll
        for (int v = 0; v < 8; ++v)
          dst[(size_t)v * KPAD] = (bf16)acc[rt][ct][v];
      }
    }
  }
}

// ---------------------------------------------------------------------------
// hidd (embedding concat) built directly transposed + bf16:
// hiddT[t][(b*112+m)][k]  -> the B^T operand of the message-passing GEMM,
// contiguous across t so all timesteps batch into one GEMM.
// ---------------------------------------------------------------------------
__global__ void embed_kernel(const float* __restrict__ hist,
                             const float* __restrict__ node_emb,
                             const float* __restrict__ tid_emb,
                             const float* __restrict__ diw_emb,
                             const float* __restrict__ W_in,
                             const float* __restrict__ b_in,
                             bf16* __restrict__ hiddT)
{
  const int idx = blockIdx.x * blockDim.x + threadIdx.x;
  if (idx >= LL * BB * NN) return;
  const int t = idx / (BB * NN);
  const int rem = idx - t * BB * NN;
  const int b = rem / NN;
  const int k = rem - b * NN;
  const float* h = hist + (((size_t)b * LL + t) * NN + k) * 3;
  const float x0 = h[0], x1 = h[1], x2 = h[2];
  int ti = (int)(x1 * 288.f); ti = min(max(ti, 0), 287);
  int di = (int)(x2 * 7.f);   di = min(max(di, 0), 6);
  bf16* dst = hiddT + ((size_t)t * CC + (size_t)b * MM) * NN + k;
  for (int m = 0; m < 64; ++m) {
    const float v = x0 * W_in[m] + x1 * W_in[64 + m] + x2 * W_in[128 + m] + b_in[m];
    dst[(size_t)m * NN] = (bf16)v;
  }
  for (int j = 0; j < 16; ++j) {
    dst[(size_t)(64 + j) * NN] = (bf16)node_emb[k * 16 + j];
    dst[(size_t)(80 + j) * NN] = (bf16)tid_emb[ti * 16 + j];
    dst[(size_t)(96 + j) * NN] = (bf16)diw_emb[di * 16 + j];
  }
}

// A_dy = softmax(relu(E E^T)) rows, emitted directly in bf16. One block/row.
__global__ __launch_bounds__(256)
void adp_adj_kernel(const float* __restrict__ E, bf16* __restrict__ Ady)
{
  __shared__ float srow[NN];
  __shared__ float sred[256];
  const int n = blockIdx.x, tid = threadIdx.x;
  float e[16];
#pragma unroll
  for (int j = 0; j < 16; ++j) e[j] = E[n * 16 + j];
  float lmax = -1e30f;
  for (int k = tid; k < NN; k += 256) {
    const float* ek = E + k * 16;
    float d = 0.f;
#pragma unroll
    for (int j = 0; j < 16; ++j) d += e[j] * ek[j];
    d = fmaxf(d, 0.f);
    srow[k] = d;
    lmax = fmaxf(lmax, d);
  }
  sred[tid] = lmax; __syncthreads();
  for (int s = 128; s > 0; s >>= 1) {
    if (tid < s) sred[tid] = fmaxf(sred[tid], sred[tid + s]);
    __syncthreads();
  }
  const float mx = sred[0]; __syncthreads();
  float lsum = 0.f;
  for (int k = tid; k < NN; k += 256) {
    const float v = __expf(srow[k] - mx);
    srow[k] = v; lsum += v;
  }
  sred[tid] = lsum; __syncthreads();
  for (int s = 128; s > 0; s >>= 1) {
    if (tid < s) sred[tid] += sred[tid + s];
    __syncthreads();
  }
  const float inv = 1.f / sred[0];
  for (int k = tid; k < NN; k += 256)
    Ady[(size_t)n * NN + k] = (bf16)(srow[k] * inv);
}

__global__ void cvt_bf16_kernel(const float* __restrict__ src, bf16* __restrict__ dst, int n)
{
  const int i = blockIdx.x * blockDim.x + threadIdx.x;
  if (i < n) dst[i] = (bf16)src[i];
}

// W[112][448] -> Wt[448][128] (transposed B^T operand, zero K-pad)
__global__ void build_wt_kernel(const float* __restrict__ W, bf16* __restrict__ Wt)
{
  const int i = blockIdx.x * blockDim.x + threadIdx.x;
  if (i >= G4M * KPAD) return;
  const int j = i / KPAD, kk = i - j * KPAD;
  Wt[i] = (kk < MM) ? (bf16)W[kk * G4M + j] : (bf16)0.f;
}

// LSTM-style cell update; only the last step's hidden state is kept.
__global__ void lstm_kernel(const float* __restrict__ gates, const float* __restrict__ bias,
                            float* __restrict__ ct, float* __restrict__ finalbuf,
                            int first, int last, int zA, int zB)
{
  const int idx = blockIdx.x * blockDim.x + threadIdx.x;
  if (idx >= BB * NN * MM) return;
  const int b = idx / (NN * MM);
  const int rem = idx - b * NN * MM;
  const int n = rem / MM;
  const int m = rem - n * MM;
  const float* g = gates + ((size_t)b * NN + n) * G4M;
  const float iv = g[m]          + bias[m];
  const float fv = g[MM + m]     + bias[MM + m];
  const float gv = g[2 * MM + m] + bias[2 * MM + m];
  const float cprev = first ? 0.f : ct[idx];
  const float c = (1.f / (1.f + __expf(-fv))) * cprev
                + (1.f / (1.f + __expf(-iv))) * tanhf(gv);
  ct[idx] = c;
  if (last) {
    const float ov = g[3 * MM + m] + bias[3 * MM + m];
    const float hval = (1.f / (1.f + __expf(-ov))) * tanhf(c);
    finalbuf[(((size_t)b * 3 + zA) * NN + n) * MM + m] = hval;
    if (zB >= 0)
      finalbuf[(((size_t)b * 3 + zB) * NN + n) * MM + m] = hval;
  }
}

// Decoder: Conv2d(layers->OUT, kernel (1,M)) then 1x1 Conv. One thread per (b,n).
__global__ void decoder_kernel(const float* __restrict__ finalbuf,
                               const float* __restrict__ W_dec,
                               const float* __restrict__ b_dec,
                               const float* __restrict__ W_out,
                               const float* __restrict__ b_out,
                               float* __restrict__ out)
{
  const int idx = blockIdx.x * blockDim.x + threadIdx.x;
  if (idx >= BB * NN) return;
  const int b = idx / NN, n = idx - (idx / NN) * NN;
  float o1[12];
#pragma unroll
  for (int o = 0; o < 12; ++o) {
    float acc = b_dec[o];
    for (int z = 0; z < 3; ++z) {
      const float* f  = finalbuf + (((size_t)b * 3 + z) * NN + n) * MM;
      const float* wd = W_dec + ((size_t)o * 3 + z) * MM;
      for (int m = 0; m < MM; ++m) acc += f[m] * wd[m];
    }
    o1[o] = acc;
  }
#pragma unroll
  for (int p = 0; p < 12; ++p) {
    float acc = b_out[p];
#pragma unroll
    for (int o = 0; o < 12; ++o) acc += W_out[p * 12 + o] * o1[o];
    out[((size_t)b * 12 + p) * NN + n] = acc;
  }
}

// ---------------------------------------------------------------------------
extern "C" void kernel_launch(void* const* d_in, const int* in_sizes, int n_in,
                              void* d_out, int out_size, void* d_ws, size_t ws_size,
                              hipStream_t stream)
{
  const float* hist     = (const float*)d_in[0];
  const float* adj      = (const float*)d_in[1];
  const float* node_emb = (const float*)d_in[2];
  const float* tid_emb  = (const float*)d_in[3];
  const float* diw_emb  = (const float*)d_in[4];
  const float* W_in     = (const float*)d_in[5];
  const float* b_in     = (const float*)d_in[6];
  const float* W_pre    = (const float*)d_in[7];
  const float* b_pre    = (const float*)d_in[8];
  const float* E_dy     = (const float*)d_in[9];
  const float* W_dy     = (const float*)d_in[10];
  const float* b_dy     = (const float*)d_in[11];
  const float* W_dec    = (const float*)d_in[12];
  const float* b_dec    = (const float*)d_in[13];
  const float* W_out    = (const float*)d_in[14];
  const float* b_out    = (const float*)d_in[15];

  char* ws = (char*)d_ws;
  size_t off = 0;
  auto carve = [&](size_t bytes) -> void* {
    void* p = ws + off;
    off += (bytes + 255) & ~(size_t)255;
    return p;
  };
  bf16*  hiddT    = (bf16*) carve((size_t)LL * CC * NN * 2);          // 84 MB
  bf16*  adjbf    = (bf16*) carve((size_t)NN * NN * 2);               // 8 MB
  bf16*  adybf    = (bf16*) carve((size_t)NN * NN * 2);               // 8 MB
  bf16*  wtpre    = (bf16*) carve((size_t)G4M * KPAD * 2);
  bf16*  wtdy     = (bf16*) carve((size_t)G4M * KPAD * 2);
  bf16*  xa_all   = (bf16*) carve((size_t)LL * BB * NN * KPAD * 2);   // 96 MB
  float* gates    = (float*)carve((size_t)BB * NN * G4M * 4);         // 56 MB
  float* ct       = (float*)carve((size_t)BB * NN * MM * 4);          // 14 MB
  float* finalbuf = (float*)carve((size_t)BB * 3 * NN * MM * 4);      // 42 MB

  // zero the K-padding columns (112..127) of the xa operand once per call
  hipMemsetAsync(xa_all, 0, (size_t)LL * BB * NN * KPAD * 2, stream);

  {
    const int n = NN * NN;
    cvt_bf16_kernel<<<(n + 255) / 256, 256, 0, stream>>>(adj, adjbf, n);
  }
  adp_adj_kernel<<<NN, 256, 0, stream>>>(E_dy, adybf);
  build_wt_kernel<<<(G4M * KPAD + 255) / 256, 256, 0, stream>>>(W_pre, wtpre);
  build_wt_kernel<<<(G4M * KPAD + 255) / 256, 256, 0, stream>>>(W_dy, wtdy);
  embed_kernel<<<(LL * BB * NN + 255) / 256, 256, 0, stream>>>(
      hist, node_emb, tid_emb, diw_emb, W_in, b_in, hiddT);

  for (int layer = 0; layer < 2; ++layer) {
    const bf16*  Ab   = layer ? adybf : adjbf;
    const bf16*  Wt   = layer ? wtdy  : wtpre;
    const float* bias = layer ? b_dy  : b_pre;
    const int zA = layer ? 1 : 0;
    const int zB = layer ? 2 : -1;      // dynamic layer result reused for z=2
    // xa[t,b,n,m] = sum_k A[n,k] * hidd[t,b,k,m] for ALL timesteps at once
    // (M=2048, N=21504, K=2048)
    dim3 g1(NN / 128, CCL / 128);
    gemm_wmma<1><<<g1, 256, 0, stream>>>(
        Ab, hiddT, nullptr, xa_all, CCL, NN, NN, NN, 0);
    for (int t = 0; t < LL; ++t) {
      // gates[(b,n),j] = sum_m xa * W[m,j]   (M=32768, N=448, K=128)
      dim3 g2((BB * NN) / 128, (G4M + 127) / 128);
      gemm_wmma<0><<<g2, 256, 0, stream>>>(
          xa_all + (size_t)t * BB * NN * KPAD, Wt, gates, nullptr,
          G4M, KPAD, KPAD, KPAD, G4M);
      lstm_kernel<<<(BB * NN * MM + 255) / 256, 256, 0, stream>>>(
          gates, bias, ct, finalbuf, t == 0, t == LL - 1, zA, zB);
    }
  }

  decoder_kernel<<<(BB * NN + 255) / 256, 256, 0, stream>>>(
      finalbuf, W_dec, b_dec, W_out, b_out, (float*)d_out);
}
